// DisenKGAT_ConvE_54271206752282
// MI455X (gfx1250) — matrix operand order repaired and checked by reading, not compiled, not run on hardware
//
#include <hip/hip_runtime.h>
#include <hip/hip_bf16.h>
#include <climits>
#include <cstddef>
#include <cstdint>

typedef __attribute__((ext_vector_type(16))) __bf16 v16bf;
typedef __attribute__((ext_vector_type(8)))  float  v8f;
typedef __attribute__((ext_vector_type(4)))  unsigned int u32x4_t;
typedef __attribute__((ext_vector_type(8)))  int i32x8_t;
typedef __attribute__((ext_vector_type(4)))  int i32x4_t;

#if __has_builtin(__builtin_amdgcn_tensor_load_to_lds) && __has_builtin(__builtin_amdgcn_s_wait_tensorcnt)
#define HAVE_TDM 1
#else
#define HAVE_TDM 0
#endif

union Frag16 { v16bf v; uint4 q[2]; };

__device__ __forceinline__ unsigned short f2bf(float f) {
  unsigned u = __float_as_uint(f);
  unsigned r = u + 0x7FFFu + ((u >> 16) & 1u);   // round-to-nearest-even
  return (unsigned short)(r >> 16);
}
__device__ __forceinline__ float bf2f(unsigned short h) {
  return __uint_as_float(((unsigned)h) << 16);
}

// monotone float <-> signed-int order encoding (for atomicMax on floats)
__device__ __forceinline__ int f2ord(float f) {
  int i = __float_as_int(f);
  return (i >= 0) ? i : (i ^ 0x7fffffff);
}
__device__ __forceinline__ float ord2f(int i) {
  int j = (i >= 0) ? i : (i ^ 0x7fffffff);
  return __int_as_float(j);
}

#define BM 64
#define BN 64
#define BKK 32
#define LDT 40   // padded LDS K-stride in halves (80 B rows -> 16B-aligned chunks)

#if HAVE_TDM
// TDM: DMA one 64-row x 32-half bf16 tile (row stride lda elems) into LDS at
// lds_off, inserting 4 DWORDs of pad after every 16 DWORDs -> 80 B rows (LDT=40).
// D# packing per CDNA5 ISA §8.3/8.4 (group0: count/lds/global/type; group1:
// data_size, pad, tensor dims (huge: operands are pre-padded), tile dims, stride).
__device__ __forceinline__ void tdm_load_tile_a(const void* gaddr, int lda_elems,
                                                unsigned lds_off)
{
  unsigned long long ga = (unsigned long long)(uintptr_t)gaddr;
  u32x4_t g0;
  g0[0] = 1u;                                              // count=1 (valid user D#)
  g0[1] = lds_off;                                         // LDS byte address
  g0[2] = (unsigned)ga;                                    // global_addr[31:0]
  g0[3] = (unsigned)((ga >> 32) & 0x01FFFFFFu) | (2u << 30); // addr[56:32], type=2
  i32x8_t g1;
  g1[0] = (int)((1u << 16) | (1u << 20) | (3u << 22) | (3u << 25));
          // data_size=2B | pad_enable | pad_interval=16DW | pad_amount=4DW
  g1[1] = (int)0xFFFF0000u;                                // tensor_dim0 lo16 (0x7FFFFFFF)
  g1[2] = (int)0xFFFF7FFFu;                                // tensor_dim0 hi / dim1 lo
  g1[3] = (int)((32u << 16) | 0x7FFFu);                    // tile_dim0=32, dim1 hi
  g1[4] = 64;                                              // tile_dim1=64 rows
  g1[5] = lda_elems;                                       // tensor_dim0_stride lo32
  g1[6] = 0;                                               // stride hi / dim1_stride
  g1[7] = 0;
  i32x4_t gz = {0, 0, 0, 0};
#if __clang_major__ >= 23
  i32x8_t gz8 = {0, 0, 0, 0, 0, 0, 0, 0};
  __builtin_amdgcn_tensor_load_to_lds(g0, g1, gz, gz, gz8, 0);
#else
  __builtin_amdgcn_tensor_load_to_lds(g0, g1, gz, gz, 0);
#endif
}
#endif

// ---------------------------------------------------------------------------
// bf16 WMMA GEMM on PRE-PADDED bf16 operands (no guards in tile loads):
//   A: Mpad x Kpad (lda); B: Kpad x Npad (ldb); C fp32 and/or Cb bf16 outputs.
// block = 128 threads (4 waves), 64x64 tile, wave = 32x32 quadrant,
// 4 x v_wmma_f32_16x16x32_bf16 per K-step; A tile via TDM when available.
// ---------------------------------------------------------------------------
__global__ void __launch_bounds__(128)
wmma_gemm_bf16(const unsigned short* __restrict__ A, int lda,
               const unsigned short* __restrict__ Bm, int ldb,
               const float* __restrict__ bias,
               float* __restrict__ C, long ldc,
               unsigned short* __restrict__ Cb, long ldcb,
               long M, int Ncols, int Kdim, int act)
{
  __shared__ __align__(16) unsigned short As[BM * LDT];
  __shared__ __align__(16) unsigned short Bs[BN * LDT];

  const int tid  = threadIdx.x;
  const int lane = tid & 31;
  const int wave = tid >> 5;
  const long rowBase = (long)blockIdx.y * BM;
  const int  colBase = blockIdx.x * BN;

  const int wm = (wave >> 1) * 32;
  const int wn = (wave & 1) * 32;

  v8f c00 = {}; v8f c01 = {}; v8f c10 = {}; v8f c11 = {};

  const int arow = lane & 15;
  const int akb  = (lane < 16) ? 0 : 8;    // A elems 0..7 = K akb.., 8..15 = K akb+16..
  const int bkb  = (lane < 16) ? 0 : 8;    // B elems 0..7 = K bkb.., 8..15 = K bkb+16..
  const int bcol = lane & 15;

  for (int k0 = 0; k0 < Kdim; k0 += BKK) {
#if HAVE_TDM
    if (wave == 0)
      tdm_load_tile_a(A + rowBase * (long)lda + k0, lda,
                      (unsigned)(uintptr_t)&As[0]);
#else
    // A tile: 64 x 32 halves, branch-free b128 loads/stores
#pragma unroll
    for (int j = 0; j < 2; ++j) {
      int c = tid + j * 128;
      int r = c >> 2;
      int col8 = (c & 3) * 8;
      const unsigned short* gp = A + (rowBase + r) * (long)lda + k0 + col8;
      uint4 v = *reinterpret_cast<const uint4*>(gp);
      *reinterpret_cast<uint4*>(&As[r * LDT + col8]) = v;
    }
#endif
    // B tile transposed into Bs[n][k]: coalesced b128 global loads, b16 scatter
#pragma unroll
    for (int j = 0; j < 2; ++j) {
      int c = tid + j * 128;               // 0..255
      int k = c >> 3;                      // 32 k rows
      int n8 = (c & 7) * 8;                // 8 chunks across 64 n
      const unsigned short* gp = Bm + (long)(k0 + k) * ldb + colBase + n8;
      uint4 v = *reinterpret_cast<const uint4*>(gp);
      const unsigned short* h = reinterpret_cast<const unsigned short*>(&v);
#pragma unroll
      for (int q = 0; q < 8; ++q)
        Bs[(n8 + q) * LDT + k] = h[q];
    }
#if HAVE_TDM
    if (wave == 0) __builtin_amdgcn_s_wait_tensorcnt(0);
#endif
    __syncthreads();

    Frag16 a0, a1, b0, b1;
    const unsigned short* ap0 = &As[(wm + arow) * LDT + akb];
    const unsigned short* ap1 = &As[(wm + 16 + arow) * LDT + akb];
    const unsigned short* bp0 = &Bs[(wn + bcol) * LDT + bkb];
    const unsigned short* bp1 = &Bs[(wn + 16 + bcol) * LDT + bkb];
    a0.q[0] = *reinterpret_cast<const uint4*>(ap0);
    a0.q[1] = *reinterpret_cast<const uint4*>(ap0 + 16);
    a1.q[0] = *reinterpret_cast<const uint4*>(ap1);
    a1.q[1] = *reinterpret_cast<const uint4*>(ap1 + 16);
    b0.q[0] = *reinterpret_cast<const uint4*>(bp0);
    b0.q[1] = *reinterpret_cast<const uint4*>(bp0 + 16);
    b1.q[0] = *reinterpret_cast<const uint4*>(bp1);
    b1.q[1] = *reinterpret_cast<const uint4*>(bp1 + 16);

    c00 = __builtin_amdgcn_wmma_f32_16x16x32_bf16(false, a0.v, false, b0.v, (short)0, c00, false, false);
    c01 = __builtin_amdgcn_wmma_f32_16x16x32_bf16(false, a0.v, false, b1.v, (short)0, c01, false, false);
    c10 = __builtin_amdgcn_wmma_f32_16x16x32_bf16(false, a1.v, false, b0.v, (short)0, c10, false, false);
    c11 = __builtin_amdgcn_wmma_f32_16x16x32_bf16(false, a1.v, false, b1.v, (short)0, c11, false, false);
    __syncthreads();
  }

  // C layout: VGPR j -> lane<16 => (M=j, N=lane); lane>=16 => (M=j+8, N=lane-16)
  const int cn    = lane & 15;
  const int rbase = (lane < 16) ? 0 : 8;
  v8f accs[4] = {c00, c01, c10, c11};
#pragma unroll
  for (int t = 0; t < 4; ++t) {
    v8f acc = accs[t];
    int mo = wm + (t >> 1) * 16;
    int no = wn + (t & 1) * 16;
#pragma unroll
    for (int j = 0; j < 8; ++j) {
      long gr = rowBase + mo + rbase + j;
      int  gc = colBase + no + cn;
      if (gr < M && gc < Ncols) {
        float v = acc[j];
        if (bias) v += bias[gc];
        if (act == 1) v = tanhf(v);
        else if (act == 2) v = fmaxf(v, 0.0f);
        if (C)  C[gr * ldc + gc] = v;
        if (Cb) Cb[gr * ldcb + gc] = f2bf(v);
      }
    }
  }
}

// ---------------------------------------------------------------------------
// Fills / converts
// ---------------------------------------------------------------------------
__global__ void fill_f32(float* p, long n, float v) {
  for (long i = (long)blockIdx.x * blockDim.x + threadIdx.x; i < n;
       i += (long)gridDim.x * blockDim.x) p[i] = v;
}
__global__ void fill_i32(int* p, long n, int v) {
  for (long i = (long)blockIdx.x * blockDim.x + threadIdx.x; i < n;
       i += (long)gridDim.x * blockDim.x) p[i] = v;
}
__global__ void convert_pad_bf16(const float* __restrict__ in,
                                 unsigned short* __restrict__ out,
                                 long rows, int cols, int ld_in,
                                 int ld_out, long rows_pad)
{
  long i = (long)blockIdx.x * 256 + threadIdx.x;
  if (i >= rows_pad * ld_out) return;
  long r = i / ld_out; int c = (int)(i % ld_out);
  float v = (r < rows && c < cols) ? in[r * ld_in + c] : 0.0f;
  out[i] = f2bf(v);
}
__global__ void gather_xi_bf16(const float* __restrict__ subemb,
                               unsigned short* __restrict__ out,
                               int Bn, int pi)
{
  long i = (long)blockIdx.x * 256 + threadIdx.x;
  if (i >= (long)Bn * 224) return;
  int b = (int)(i / 224), c = (int)(i % 224);
  float v = (c < 200) ? subemb[(long)b * 600 + pi * 200 + c] : 0.0f;
  out[i] = f2bf(v);
}

// ---------------------------------------------------------------------------
// Edge pass 1 (bf16 features): logit + ordered-int atomicMax segment max
// ---------------------------------------------------------------------------
__global__ void __launch_bounds__(256)
edge_logit(const int* __restrict__ ei, const int* __restrict__ et,
           const unsigned short* __restrict__ x0, const float* __restrict__ init_rel,
           const float* __restrict__ rel_weight,
           float* __restrict__ logit, int* __restrict__ smax, int E)
{
  int e = blockIdx.x * 8 + (threadIdx.x >> 5);
  int lane = threadIdx.x & 31;
  if (e >= E) return;
  int dst = ei[e], src = ei[E + e], t = et[e];
  const unsigned short* xd = x0 + (long)dst * 600;
  const unsigned short* xs = x0 + (long)src * 600;
  const float* re = init_rel + (long)t * 200;
  __builtin_prefetch(re, 0, 3);          // global_prefetch_b8
#pragma unroll
  for (int k = 0; k < 3; ++k) {
    float partial = 0.0f;
    for (int d = lane * 2; d < 200; d += 64) {
      unsigned pd = *reinterpret_cast<const unsigned*>(xd + k * 200 + d);
      unsigned ps = *reinterpret_cast<const unsigned*>(xs + k * 200 + d);
      partial += bf2f((unsigned short)pd) * bf2f((unsigned short)ps) * re[d];
      partial += bf2f((unsigned short)(pd >> 16)) * bf2f((unsigned short)(ps >> 16)) * re[d + 1];
    }
    for (int off = 16; off > 0; off >>= 1)
      partial += __shfl_down(partial, off, 32);
    if (lane == 0) {
      float l = partial * rel_weight[t * 3 + k];
      l = (l > 0.0f) ? l : 0.2f * l;     // leaky_relu 0.2
      logit[e * 3 + k] = l;
      atomicMax(&smax[dst * 3 + k], f2ord(l));
    }
  }
}

__global__ void __launch_bounds__(256)
edge_exp(const int* __restrict__ ei, const float* __restrict__ logit_in,
         const int* __restrict__ smax, float* __restrict__ logit_out,
         float* __restrict__ denom, int E)
{
  long i = (long)blockIdx.x * 256 + threadIdx.x;
  if (i >= (long)E * 3) return;
  int e = (int)(i / 3), k = (int)(i % 3);
  int dst = ei[e];
  int enc = smax[dst * 3 + k];
  float m = (enc == INT_MIN) ? 0.0f : ord2f(enc);
  if (!isfinite(m)) m = 0.0f;
  float a = __expf(logit_in[i] - m);
  logit_out[i] = a;
  atomicAdd(&denom[dst * 3 + k], a);
}

// agg (120 MB fp32) fits the 192 MB L2 -> atomics resolve in L2.
__global__ void __launch_bounds__(256)
edge_agg(const int* __restrict__ ei, const int* __restrict__ et,
         const unsigned short* __restrict__ x0, const float* __restrict__ init_rel,
         const float* __restrict__ a_buf, const float* __restrict__ denom,
         float* __restrict__ agg, int E)
{
  int e = blockIdx.x * 8 + (threadIdx.x >> 5);
  int lane = threadIdx.x & 31;
  if (e >= E) return;
  int dst = ei[e], src = ei[E + e], t = et[e];
  const unsigned short* xs = x0 + (long)src * 600;
  const float* re = init_rel + (long)t * 200;
  float* ag = agg + (long)dst * 600;
  __builtin_prefetch(xs, 0, 3);
#pragma unroll
  for (int k = 0; k < 3; ++k) {
    float alpha = a_buf[e * 3 + k] / (denom[dst * 3 + k] + 1e-10f);
    for (int d = lane * 2; d < 200; d += 64) {
      unsigned ps = *reinterpret_cast<const unsigned*>(xs + k * 200 + d);
      atomicAdd(&ag[k * 200 + d],     alpha * bf2f((unsigned short)ps) * re[d]);
      atomicAdd(&ag[k * 200 + d + 1], alpha * bf2f((unsigned short)(ps >> 16)) * re[d + 1]);
    }
  }
}

__global__ void gather_rows(const float* __restrict__ src, const int* __restrict__ idx,
                            float* __restrict__ dst, long rows, int width)
{
  long i = (long)blockIdx.x * 256 + threadIdx.x;
  if (i >= rows * width) return;
  long r = i / width; int c = (int)(i % width);
  dst[i] = src[(long)idx[r] * width + c];
}

__global__ void __launch_bounds__(256)
club_reduce(const float* __restrict__ mu, const float* __restrict__ lv,
            const float* __restrict__ subemb, const int* __restrict__ perm,
            int pj, int Bn, float scale, float* __restrict__ out_mi)
{
  long i = (long)blockIdx.x * 256 + threadIdx.x;
  float v = 0.0f;
  if (i < (long)Bn * 200) {
    int b = (int)(i / 200), d = (int)(i % 200);
    float m  = mu[i];
    float iv = __expf(-lv[i]);
    float y  = subemb[(long)b * 600 + pj * 200 + d];
    float yn = subemb[(long)perm[b] * 600 + pj * 200 + d];
    float dp = m - y, dn = m - yn;
    v = (dn * dn - dp * dp) * iv;   // pos - neg
  }
  __shared__ float red[256];
  red[threadIdx.x] = v;
  __syncthreads();
  for (int s = 128; s > 0; s >>= 1) {
    if (threadIdx.x < s) red[threadIdx.x] += red[threadIdx.x + s];
    __syncthreads();
  }
  if (threadIdx.x == 0) atomicAdd(out_mi, red[0] * scale);
}

// ---------------------------------------------------------------------------
extern "C" void kernel_launch(void* const* d_in, const int* in_sizes, int n_in,
                              void* d_out, int out_size, void* d_ws, size_t ws_size,
                              hipStream_t stream)
{
  const int*   edge_index = (const int*)d_in[0];
  const int*   edge_type  = (const int*)d_in[1];
  const int*   sub        = (const int*)d_in[2];
  const int*   rel        = (const int*)d_in[3];
  const int*   perm       = (const int*)d_in[4];
  const float* init_embed = (const float*)d_in[5];
  const float* init_rel   = (const float*)d_in[6];
  const float* pca_W      = (const float*)d_in[7];
  const float* pca_b      = (const float*)d_in[8];
  const float* rel_weight = (const float*)d_in[9];
  const float* W_ent      = (const float*)d_in[10];
  // d_in[11] = W_rel : dead (r_new never reaches the outputs)
  const float* mu_w1 = (const float*)d_in[12];
  const float* mu_b1 = (const float*)d_in[13];
  const float* mu_w2 = (const float*)d_in[14];
  const float* mu_b2 = (const float*)d_in[15];
  const float* lv_w1 = (const float*)d_in[16];
  const float* lv_b1 = (const float*)d_in[17];
  const float* lv_w2 = (const float*)d_in[18];
  const float* lv_b2 = (const float*)d_in[19];

  const int D = 200, K = 3, H = 100;
  const int Dp = 224, Hp = 128;                 // K-dim pads (mult of 32)
  const int E  = in_sizes[1];
  const int Bb = in_sizes[2];
  const long Nn = in_sizes[5] / D;              // 50000
  const int KD = K * D;                         // 600
  const long NnP  = (Nn + 63) & ~63L;           // 50048
  const long NKP  = (Nn * K + 63) & ~63L;       // 150016

  // ---- workspace layout: fp32 region, then 16B-aligned bf16 region ----
  float* aggf  = (float*)d_ws;                          // Nn*KD
  float* lg    = aggf + (size_t)Nn * KD;                // E*K
  float* denom = lg + (size_t)E * K;                    // Nn*K
  float* mu    = denom + (size_t)Nn * K;                // Bb*D
  float* lv    = mu + (size_t)Bb * D;                   // Bb*D
  int*   smax  = (int*)(lv + (size_t)Bb * D);           // Nn*K
  unsigned short* hb =
      (unsigned short*)(((uintptr_t)(smax + (size_t)Nn * K) + 15) & ~(uintptr_t)15);
  unsigned short* Aebf   = hb;                          // NnP * Dp
  unsigned short* pcaWbf = Aebf   + (size_t)NnP * Dp;   // Dp * 640
  unsigned short* Wentbf = pcaWbf + (size_t)Dp * 640;   // Dp * 256
  unsigned short* aggbf  = Wentbf + (size_t)Dp * 256;   // NKP * Dp
  unsigned short* x0bf   = aggbf  + (size_t)NKP * Dp;   // Nn * KD
  unsigned short* xibf   = x0bf   + (size_t)Nn * KD;    // Bb * Dp
  unsigned short* h1bf   = xibf   + (size_t)Bb * Dp;    // Bb * Hp
  unsigned short* w1bf   = h1bf   + (size_t)Bb * Hp;    // 6 * Dp*Hp
  unsigned short* w2bf   = w1bf   + (size_t)6 * Dp * Hp;// 6 * Hp*256

  // ---- output layout ----
  float* out_sub = (float*)d_out;                       // Bb*KD
  float* out_rel = out_sub + (size_t)Bb * KD;           // Bb*D
  float* out_x   = out_rel + (size_t)Bb * D;            // Nn*KD
  float* out_mi  = out_x + (size_t)Nn * KD;             // 1

  auto gemm = [&](const unsigned short* A, int lda, const unsigned short* Bm, int ldb,
                  const float* bias, float* C, long ldc,
                  unsigned short* Cb, long ldcb,
                  long M, int Ncols, int Kdim, int act) {
    dim3 g((unsigned)((Ncols + BN - 1) / BN), (unsigned)((M + BM - 1) / BM));
    wmma_gemm_bf16<<<g, 128, 0, stream>>>(A, lda, Bm, ldb, bias, C, ldc, Cb, ldcb,
                                          M, Ncols, Kdim, act);
  };
  auto conv = [&](const float* in, unsigned short* out, long rows, int cols,
                  int ld_in, int ld_out, long rows_pad) {
    long n = rows_pad * ld_out;
    convert_pad_bf16<<<(unsigned)((n + 255) / 256), 256, 0, stream>>>(
        in, out, rows, cols, ld_in, ld_out, rows_pad);
  };

  // ---- init accumulators ----
  fill_f32<<<4096, 256, 0, stream>>>(aggf, Nn * KD, 0.0f);
  fill_i32<<<(unsigned)((Nn * K + 255) / 256), 256, 0, stream>>>(smax, Nn * K, INT_MIN);
  fill_f32<<<(unsigned)((Nn * K + 255) / 256), 256, 0, stream>>>(denom, Nn * K, 0.0f);
  fill_f32<<<1, 256, 0, stream>>>(out_mi, 1, 0.0f);
  fill_i32<<<(unsigned)(((long)Bb * Hp / 2 + 255) / 256), 256, 0, stream>>>(
      (int*)h1bf, (long)Bb * Hp / 2, 0);

  // ---- one-time bf16 operand conversion ----
  conv(init_embed, Aebf, Nn, D, D, Dp, NnP);
  conv(pca_W, pcaWbf, D, KD, KD, 640, Dp);
  conv(W_ent, Wentbf, D, D, D, 256, Dp);
  for (int p = 0; p < 3; ++p) {
    conv(mu_w1 + (size_t)p * D * H, w1bf + (size_t)p * Dp * Hp,       D, H, H, Hp, Dp);
    conv(lv_w1 + (size_t)p * D * H, w1bf + (size_t)(3 + p) * Dp * Hp, D, H, H, Hp, Dp);
    conv(mu_w2 + (size_t)p * H * D, w2bf + (size_t)p * Hp * 256,       H, D, D, 256, Hp);
    conv(lv_w2 + (size_t)p * H * D, w2bf + (size_t)(3 + p) * Hp * 256, H, D, D, 256, Hp);
  }

  // 1) x0 = tanh(init_embed @ pca_W + pca_b)  -> bf16 features
  gemm(Aebf, Dp, pcaWbf, 640, pca_b, nullptr, 0, x0bf, KD, Nn, KD, D, /*tanh*/1);

  // 2) edge attention + aggregation (bandwidth stage, bf16 reads, L2 atomics)
  edge_logit<<<(unsigned)((E + 7) / 8), 256, 0, stream>>>(
      edge_index, edge_type, x0bf, init_rel, rel_weight, lg, smax, E);
  edge_exp<<<(unsigned)(((long)E * K + 255) / 256), 256, 0, stream>>>(
      edge_index, lg, smax, lg, denom, E);
  edge_agg<<<(unsigned)((E + 7) / 8), 256, 0, stream>>>(
      edge_index, edge_type, x0bf, init_rel, lg, denom, aggf, E);

  // 3) x_new = tanh(agg @ W_ent)
  conv(aggf, aggbf, Nn * K, D, D, Dp, NKP);
  gemm(aggbf, Dp, Wentbf, 256, nullptr, out_x, D, nullptr, 0, Nn * K, D, D, /*tanh*/1);

  // 4) gathers
  gather_rows<<<(unsigned)(((long)Bb * KD + 255) / 256), 256, 0, stream>>>(
      out_x, sub, out_sub, Bb, KD);
  gather_rows<<<(unsigned)(((long)Bb * D + 255) / 256), 256, 0, stream>>>(
      init_rel, rel, out_rel, Bb, D);

  // 5) CLUB MI bound: 3 pairs x 4 small WMMA GEMMs + reduction
  static const int pi_[3] = {0, 0, 1};
  static const int pj_[3] = {1, 2, 2};
  const float scale = 0.5f / (float)Bb;
  for (int p = 0; p < 3; ++p) {
    gather_xi_bf16<<<(unsigned)(((long)Bb * Dp + 255) / 256), 256, 0, stream>>>(
        out_sub, xibf, Bb, pi_[p]);
    gemm(xibf, Dp, w1bf + (size_t)p * Dp * Hp, Hp, mu_b1 + (size_t)p * H,
         nullptr, 0, h1bf, Hp, Bb, H, D, /*relu*/2);
    gemm(h1bf, Hp, w2bf + (size_t)p * Hp * 256, 256, mu_b2 + (size_t)p * D,
         mu, D, nullptr, 0, Bb, D, H, /*none*/0);
    gemm(xibf, Dp, w1bf + (size_t)(3 + p) * Dp * Hp, Hp, lv_b1 + (size_t)p * H,
         nullptr, 0, h1bf, Hp, Bb, H, D, /*relu*/2);
    gemm(h1bf, Hp, w2bf + (size_t)(3 + p) * Hp * 256, 256, lv_b2 + (size_t)p * D,
         lv, D, nullptr, 0, Bb, D, H, /*tanh*/1);
    club_reduce<<<(unsigned)(((long)Bb * D + 255) / 256), 256, 0, stream>>>(
        mu, lv, out_sub, perm, pj_[p], Bb, scale, out_mi);
  }
}
